// SpMiddleResNetFHDPruningBase_19104014532898
// MI455X (gfx1250) — matrix operand order repaired and checked by reading, not compiled, not run on hardware
//
#include <hip/hip_runtime.h>

// ---------------------------------------------------------------------------
// SpMiddleResNetFHD dense-grid backbone for MI455X (gfx1250, wave32, WMMA).
//
// v3: conv kernel templated on the FULL conv geometry (channels, kernel,
// dims, strides, pads) + halo-padded f16 activation buffers. The K loop is
// fully unrolled: every tap/channel decomposition constant-folds, halo
// padding removes all bounds checks, W-edge A rows are clamped (stores
// skipped), so the steady state is pure global_load_b128 (immediate offsets)
// + v_wmma_f32_16x16x32_f16. B operand is pre-swizzled into WMMA fragment
// order; each wave does 2 cout-groups (COUT>=32) to reuse the A fragment.
// Net is HBM-bound (~2 GB traffic vs ~475 GFLOP) so per-WMMA instruction
// overhead is the thing to minimize.
// ---------------------------------------------------------------------------

typedef __attribute__((ext_vector_type(16))) _Float16 hvec16;
typedef __attribute__((ext_vector_type(8)))  _Float16 hvec8;
typedef __attribute__((ext_vector_type(8)))  float    fvec8;

#define EPS_BN 1e-3f

// Stage geometry (fixed for this workload)
constexpr int D1 = 41, H1 = 304, W1 = 304;
constexpr int D2 = 21, H2 = 152, W2 = 152;
constexpr int D3 = 11, H3 = 76,  W3 = 76;
constexpr int D4 = 5,  H4 = 38,  W4 = 38;
constexpr int De = 2,  He = 38,  We = 38;

struct ConvPtrs {
  const _Float16* in;     // halo-padded [DIN+2,HIN+2,WIN+2,CIN] f16
  const _Float16* wts;    // WMMA-B swizzled: [cog][chunk][lane32][16] f16
  const _Float16* resid;  // halo-padded f16 or null
  const float* bnb; const float* bng; const float* bnm; const float* bnv;
  const float* bias;      // [COUT] or null
  const float* mask;      // dense [DOUT,HOUT,WOUT] f32 (0/1)
  _Float16* out16;        // halo-padded f16 or null
  float* out32;           // dense f32 output slice or null
};

template <int CIN, int COUT, int KD, int KH, int KW,
          int DIN, int HIN, int WIN, int DOUT, int HOUT, int WOUT,
          int SD, int SH, int SW, int PD, int PH, int PW, int LAYOUT>
__global__ __launch_bounds__(128) void conv3d_wmma(ConvPtrs p) {
  constexpr int KHW   = KH * KW;
  constexpr int KTOT  = KD * KHW * CIN;
  constexpr int NCH   = (KTOT + 31) / 32;      // K chunks of 32
  constexpr int NT    = (COUT >= 32) ? 2 : 1;  // cout-groups per wave
  constexpr int CG    = COUT / (16 * NT);
  constexpr int WT    = (WOUT + 15) / 16;
  constexpr long long TOTAL = (long long)DOUT * HOUT * WT * CG;
  constexpr int HINH  = HIN + 2,  WINH  = WIN + 2;   // halo strides
  constexpr int HOUTH = HOUT + 2, WOUTH = WOUT + 2;

  const int lane = threadIdx.x & 31;
  const long long wave =
      (long long)blockIdx.x * (blockDim.x >> 5) + (threadIdx.x >> 5);
  if (wave >= TOTAL) return;  // wave-uniform: EXEC all-ones for WMMA

  long long t = wave;
  const int cog0 = ((int)(t % CG)) * NT; t /= CG;
  const int wb   = ((int)(t % WT)) << 4; t /= WT;
  const int ho   = (int)(t % HOUT); t /= HOUT;
  const int dd   = (int)t;

  const int half = lane >> 4;
  const int col  = lane & 15;

  // A row voxel along W, clamped so edge tiles need no bounds checks at all
  // (duplicate rows compute garbage that the epilogue never stores).
  int wo_a = wb + col;
  if (wo_a > WOUT - 1) wo_a = WOUT - 1;

  // Halo-space base address of this row's (dz=0,dy=0,dx=0,ci=0) tap element.
  const long long aOff =
      (((long long)(dd * SD - PD + 1) * HINH + (ho * SH - PH + 1)) * WINH +
       (wo_a * SW - PW + 1)) * CIN;
  // For CIN>=16 the lane-half's +8 channel offset is constant -> fold into base.
  const _Float16* aB = p.in + aOff + ((CIN >= 16) ? (half << 3) : 0);
  const _Float16* wB = p.wts + (long long)cog0 * NCH * 512 + lane * 16;

  // compile-time tap offset for K index k (relative to aOff, halo space)
  auto koff = [](int k) -> long long {
    const int tap = k / CIN, ci = k % CIN;
    const int dz = tap / KHW, rr = tap % KHW;
    const int dy = rr / KW, dx = rr % KW;
    return (long long)((dz * HINH + dy) * WINH + dx) * CIN + ci;
  };

  const fvec8 zero = {};
  fvec8 acc[NT];
#pragma unroll
  for (int i = 0; i < NT; ++i) acc[i] = zero;

#pragma unroll
  for (int ch = 0; ch < NCH; ++ch) {
    hvec16 A;
#pragma unroll
    for (int r = 0; r < 2; ++r) {
      hvec8 run = {};
      const int k0 = (ch << 5) + (r << 4);  // constant after unroll
      if constexpr (CIN >= 16) {
        // ci0 is a multiple of 16 -> both halves stay inside one tap;
        // validity uniform; offset is a compile-time immediate.
        if (k0 < KTOT) run = *(const hvec8*)(aB + koff(k0));
      } else {
        // CIN==8: the two halves hit adjacent taps -> constant-pair select.
        const int kh = k0 + (half << 3);
        if (kh < KTOT) {
          const long long o0 = koff(k0);
          const long long o1 = koff((k0 + 8 < KTOT) ? (k0 + 8) : 0);
          run = *(const hvec8*)(aB + (half ? o1 : o0));
        }
      }
#pragma unroll
      for (int j = 0; j < 8; ++j) A[(r << 3) + j] = run[j];
    }

#pragma unroll
    for (int nt = 0; nt < NT; ++nt) {
      const _Float16* wp = wB + ((long long)nt * NCH + ch) * 512;
      const hvec8 b0 = *(const hvec8*)(wp);
      const hvec8 b1 = *(const hvec8*)(wp + 8);
      hvec16 B;
#pragma unroll
      for (int j = 0; j < 8; ++j) { B[j] = b0[j]; B[8 + j] = b1[j]; }
      acc[nt] = __builtin_amdgcn_wmma_f32_16x16x32_f16(false, A, false, B,
                                                       (short)0, acc[nt],
                                                       false, false);
    }
  }

  // ---- epilogue: folded bias+BN, residual, ReLU, active-site mask ----
#pragma unroll
  for (int nt = 0; nt < NT; ++nt) {
    const int co = (cog0 + nt) * 16 + col;
    const float sc = p.bng[co] * rsqrtf(p.bnv[co] + EPS_BN);
    const float bs = p.bias ? p.bias[co] : 0.0f;
    const float sh = (bs - p.bnm[co]) * sc + p.bnb[co];
#pragma unroll
    for (int r = 0; r < 8; ++r) {
      const int wo = wb + r + (half << 3);  // D row -> output voxel along W
      if (wo >= WOUT) continue;
      const long long voxD = ((long long)dd * HOUT + ho) * WOUT + wo;
      const long long voxH =
          ((long long)(dd + 1) * HOUTH + (ho + 1)) * WOUTH + (wo + 1);
      float y = acc[nt][r] * sc + sh;
      if (p.resid) y += (float)p.resid[voxH * COUT + co];
      y = fmaxf(y, 0.0f) * p.mask[voxD];
      if (p.out16) p.out16[voxH * COUT + co] = (_Float16)y;
      if (p.out32) {
        const long long o =
            (LAYOUT == 0)
                ? voxD * COUT + co
                : (((long long)co * DOUT + dd) * HOUT + ho) * WOUT + wo;
        p.out32[o] = y;
      }
    }
  }
}

// ---- active-site mask propagation for strided SparseConv3d ----------------
struct MaskP {
  const float* min_;
  float* mout;
  int Din, Hin, Win, Dout, Hout, Wout;
  int kd, kh, kw, sd, sh, sw, pd, ph, pw;
};

__global__ void mask_down_k(MaskP p) {
  const long long tot = (long long)p.Dout * p.Hout * p.Wout;
  const long long i = (long long)blockIdx.x * blockDim.x + threadIdx.x;
  if (i >= tot) return;
  const int wo = (int)(i % p.Wout);
  long long t = i / p.Wout;
  const int ho = (int)(t % p.Hout);
  const int dd = (int)(t / p.Hout);
  float r = 0.0f;
  for (int dz = 0; dz < p.kd; ++dz) {
    const int di = dd * p.sd - p.pd + dz;
    if ((unsigned)di >= (unsigned)p.Din) continue;
    for (int dy = 0; dy < p.kh; ++dy) {
      const int hi = ho * p.sh - p.ph + dy;
      if ((unsigned)hi >= (unsigned)p.Hin) continue;
      for (int dx = 0; dx < p.kw; ++dx) {
        const int wi = wo * p.sw - p.pw + dx;
        if ((unsigned)wi >= (unsigned)p.Win) continue;
        if (p.min_[((long long)di * p.Hin + hi) * p.Win + wi] > 0.0f) r = 1.0f;
      }
    }
  }
  p.mout[i] = r;
}

// ---- scatter voxel features into halo-padded dense f16 grid ---------------
__global__ void scatter_k(const float* vf, const int* coors, int N,
                          _Float16* dense8, float* mask1, int D, int H, int W) {
  const int i = blockIdx.x * blockDim.x + threadIdx.x;
  if (i >= N) return;
  const int z = coors[i * 4 + 1], y = coors[i * 4 + 2], x = coors[i * 4 + 3];
  if ((unsigned)z >= (unsigned)D || (unsigned)y >= (unsigned)H ||
      (unsigned)x >= (unsigned)W)
    return;
  const long long vH = ((long long)(z + 1) * (H + 2) + (y + 1)) * (W + 2) + (x + 1);
  const long long vD = (((long long)z * H) + y) * W + x;
#pragma unroll
  for (int c = 0; c < 5; ++c) dense8[vH * 8 + c] = (_Float16)vf[i * 5 + c];
  mask1[vD] = 1.0f;
}

// ---- weight f32 [kd,kh,kw,Cin,Cout] -> f16 WMMA-B fragment layout ---------
__global__ void wprep_k(const float* src, _Float16* dst, long long total,
                        int Cin, int Cout, int CinPad, int nchunks) {
  const long long i = (long long)blockIdx.x * blockDim.x + threadIdx.x;
  if (i >= total) return;
  const int co = (int)(i % Cout);
  long long t = i / Cout;
  const int ci = (int)(t % Cin);
  const long long tap = t / Cin;
  const long long k = tap * CinPad + ci;
  const int chunk = (int)(k >> 5);
  const int lane = ((int)((k >> 3) & 1)) * 16 + (co & 15);
  const int jh = (int)(((k >> 4) & 1) * 8 + (k & 7));
  const int cog = co >> 4;
  dst[(((long long)cog * nchunks + chunk) * 32 + lane) * 16 + jh] =
      (_Float16)src[i];
}

// ---------------------------------------------------------------------------
template <int CIN, int COUT, int KD, int KH, int KW,
          int DIN, int HIN, int WIN, int DOUT, int HOUT, int WOUT,
          int SD, int SH, int SW, int PD, int PH, int PW, int LAYOUT>
static inline void run_conv(hipStream_t stream, const ConvPtrs& p) {
  constexpr int NT = (COUT >= 32) ? 2 : 1;
  constexpr int CG = COUT / (16 * NT);
  constexpr int WT = (WOUT + 15) / 16;
  constexpr long long TOTAL = (long long)DOUT * HOUT * WT * CG;
  constexpr int GRID = (int)((TOTAL + 3) / 4);  // 4 waves (128 thr) / block
  conv3d_wmma<CIN, COUT, KD, KH, KW, DIN, HIN, WIN, DOUT, HOUT, WOUT,
              SD, SH, SW, PD, PH, PW, LAYOUT><<<GRID, 128, 0, stream>>>(p);
}

static inline ConvPtrs mkp(const _Float16* in, const _Float16* wts,
                           const _Float16* resid, void* const* d_in, int bnB,
                           const float* bias, const float* mask,
                           _Float16* o16, float* o32) {
  ConvPtrs p;
  p.in = in; p.wts = wts; p.resid = resid;
  p.bnb = (const float*)d_in[bnB + 0];
  p.bng = (const float*)d_in[bnB + 1];
  p.bnm = (const float*)d_in[bnB + 2];
  p.bnv = (const float*)d_in[bnB + 3];
  p.bias = bias; p.mask = mask; p.out16 = o16; p.out32 = o32;
  return p;
}

// SparseBasicBlock: conv1 x->tmp, conv2 tmp->x (in-place, residual = x)
template <int C, int D, int H, int W>
static void resblock(void* const* d_in, hipStream_t stream, int bse,
                     _Float16* x, _Float16* tmp, const _Float16* w1,
                     const _Float16* w2, const float* mask, float* o32) {
  run_conv<C, C, 3, 3, 3, D, H, W, D, H, W, 1, 1, 1, 1, 1, 1, 0>(
      stream, mkp(x, w1, nullptr, d_in, bse + 2,
                  (const float*)d_in[bse + 0], mask, tmp, nullptr));
  run_conv<C, C, 3, 3, 3, D, H, W, D, H, W, 1, 1, 1, 1, 1, 1, 0>(
      stream, mkp(tmp, w2, x, d_in, bse + 6,
                  (const float*)d_in[bse + 1], mask, x, o32));
}

extern "C" void kernel_launch(void* const* d_in, const int* in_sizes, int n_in,
                              void* d_out, int out_size, void* d_ws,
                              size_t ws_size, hipStream_t stream) {
  (void)n_in; (void)out_size; (void)ws_size;
  const int N = in_sizes[0] / 5;

  const long long V1 = (long long)D1 * H1 * W1;
  const long long V2 = (long long)D2 * H2 * W2;
  const long long V3 = (long long)D3 * H3 * W3;
  const long long V4 = (long long)D4 * H4 * W4;
  const long long Ve = (long long)De * He * We;

  // ---- workspace carve-up (bump allocator, 256B aligned) ----
  char* base = (char*)d_ws;
  size_t off = 0;
  auto carve = [&](size_t bytes) -> void* {
    void* p = base + off;
    off = (off + bytes + 255) & ~(size_t)255;
    return p;
  };
  auto wbytes = [](int taps, int cinp, int cout) -> size_t {
    return (size_t)(cout / 16) * ((taps * cinp + 31) / 32) * 512 * 2;
  };
  auto hbytes = [](int D, int H, int W, int C) -> size_t {  // halo f16 buffer
    return (size_t)(D + 2) * (H + 2) * (W + 2) * C * 2;
  };

  _Float16* Winp = (_Float16*)carve(wbytes(27, 8, 16));
  _Float16* Wc1w[4]; for (int i = 0; i < 4; ++i) Wc1w[i] = (_Float16*)carve(wbytes(27, 16, 16));
  _Float16* Wc2d = (_Float16*)carve(wbytes(27, 16, 32));
  _Float16* Wc2w[4]; for (int i = 0; i < 4; ++i) Wc2w[i] = (_Float16*)carve(wbytes(27, 32, 32));
  _Float16* Wc3d = (_Float16*)carve(wbytes(27, 32, 64));
  _Float16* Wc3w[4]; for (int i = 0; i < 4; ++i) Wc3w[i] = (_Float16*)carve(wbytes(27, 64, 64));
  _Float16* Wc4d = (_Float16*)carve(wbytes(27, 64, 128));
  _Float16* Wc4w[4]; for (int i = 0; i < 4; ++i) Wc4w[i] = (_Float16*)carve(wbytes(27, 128, 128));
  _Float16* Wex = (_Float16*)carve(wbytes(3, 128, 128));

  _Float16* dense8 = (_Float16*)carve(hbytes(D1, H1, W1, 8));
  _Float16* bufA  = (_Float16*)carve(hbytes(D1, H1, W1, 16));
  _Float16* bufB  = (_Float16*)carve(hbytes(D1, H1, W1, 16));
  _Float16* buf2A = (_Float16*)carve(hbytes(D2, H2, W2, 32));
  _Float16* buf2B = (_Float16*)carve(hbytes(D2, H2, W2, 32));
  _Float16* buf3A = (_Float16*)carve(hbytes(D3, H3, W3, 64));
  _Float16* buf3B = (_Float16*)carve(hbytes(D3, H3, W3, 64));
  _Float16* buf4A = (_Float16*)carve(hbytes(D4, H4, W4, 128));
  _Float16* buf4B = (_Float16*)carve(hbytes(D4, H4, W4, 128));
  float* mask1  = (float*)carve((size_t)V1 * 4);
  float* mask2  = (float*)carve((size_t)V2 * 4);
  float* mask3  = (float*)carve((size_t)V3 * 4);
  float* mask4  = (float*)carve((size_t)V4 * 4);
  float* maskex = (float*)carve((size_t)Ve * 4);

  // Zero everything: weight-padding chunks, halo shells, scatter targets.
  // (~440 MB @ 23.3 TB/s ~= 19 us; interior cells are overwritten by convs.)
  hipMemsetAsync(d_ws, 0, off, stream);

  // ---- weight conversion (param leaves per sorted-key dict flattening) ----
  struct WJ { int src; _Float16* dst; int taps, cin, cinp, cout; };
  const WJ wj[] = {
      {124, Winp, 27, 5, 8, 16},
      {14, Wc1w[0], 27, 16, 16, 16}, {15, Wc1w[1], 27, 16, 16, 16},
      {26, Wc1w[2], 27, 16, 16, 16}, {27, Wc1w[3], 27, 16, 16, 16},
      {32, Wc2d, 27, 16, 16, 32},
      {43, Wc2w[0], 27, 32, 32, 32}, {44, Wc2w[1], 27, 32, 32, 32},
      {55, Wc2w[2], 27, 32, 32, 32}, {56, Wc2w[3], 27, 32, 32, 32},
      {61, Wc3d, 27, 32, 32, 64},
      {72, Wc3w[0], 27, 64, 64, 64}, {73, Wc3w[1], 27, 64, 64, 64},
      {84, Wc3w[2], 27, 64, 64, 64}, {85, Wc3w[3], 27, 64, 64, 64},
      {90, Wc4d, 27, 64, 64, 128},
      {101, Wc4w[0], 27, 128, 128, 128}, {102, Wc4w[1], 27, 128, 128, 128},
      {113, Wc4w[2], 27, 128, 128, 128}, {114, Wc4w[3], 27, 128, 128, 128},
      {119, Wex, 3, 128, 128, 128},
  };
  for (const WJ& j : wj) {
    const long long tot = (long long)j.taps * j.cin * j.cout;
    const int nch = (j.taps * j.cinp + 31) / 32;
    wprep_k<<<(int)((tot + 255) / 256), 256, 0, stream>>>(
        (const float*)d_in[j.src], j.dst, tot, j.cin, j.cout, j.cinp, nch);
  }

  scatter_k<<<(N + 255) / 256, 256, 0, stream>>>(
      (const float*)d_in[0], (const int*)d_in[1], N, dense8, mask1, D1, H1, W1);

  auto maskdown = [&](const float* mi, int Din, int Hin, int Win, float* mo,
                      int Dout, int Hout, int Wout, int kd, int kh, int kw,
                      int sd, int sh, int sw, int pd, int ph, int pw) {
    MaskP p;
    p.min_ = mi; p.mout = mo;
    p.Din = Din; p.Hin = Hin; p.Win = Win;
    p.Dout = Dout; p.Hout = Hout; p.Wout = Wout;
    p.kd = kd; p.kh = kh; p.kw = kw;
    p.sd = sd; p.sh = sh; p.sw = sw;
    p.pd = pd; p.ph = ph; p.pw = pw;
    const long long tot = (long long)Dout * Hout * Wout;
    mask_down_k<<<(int)((tot + 255) / 256), 256, 0, stream>>>(p);
  };

  float* retOut = (float*)d_out;                 // 256*38*38
  float* x1Out = retOut + 369664LL;              // V1*16
  float* x2Out = x1Out + V1 * 16;                // V2*32
  float* x3Out = x2Out + V2 * 32;                // V3*64
  float* x4Out = x3Out + V3 * 64;                // V4*128

  // conv_input: SubMConv3d(5->16) + BN + ReLU (Cin padded to 8)
  run_conv<8, 16, 3, 3, 3, D1, H1, W1, D1, H1, W1, 1, 1, 1, 1, 1, 1, 0>(
      stream, mkp(dense8, Winp, nullptr, d_in, 120, nullptr, mask1, bufA, nullptr));
  resblock<16, D1, H1, W1>(d_in, stream, 4,  bufA, bufB, Wc1w[0], Wc1w[1], mask1, nullptr);
  resblock<16, D1, H1, W1>(d_in, stream, 16, bufA, bufB, Wc1w[2], Wc1w[3], mask1, x1Out);

  maskdown(mask1, D1, H1, W1, mask2, D2, H2, W2, 3, 3, 3, 2, 2, 2, 1, 1, 1);
  run_conv<16, 32, 3, 3, 3, D1, H1, W1, D2, H2, W2, 2, 2, 2, 1, 1, 1, 0>(
      stream, mkp(bufA, Wc2d, nullptr, d_in, 28, nullptr, mask2, buf2A, nullptr));
  resblock<32, D2, H2, W2>(d_in, stream, 33, buf2A, buf2B, Wc2w[0], Wc2w[1], mask2, nullptr);
  resblock<32, D2, H2, W2>(d_in, stream, 45, buf2A, buf2B, Wc2w[2], Wc2w[3], mask2, x2Out);

  maskdown(mask2, D2, H2, W2, mask3, D3, H3, W3, 3, 3, 3, 2, 2, 2, 1, 1, 1);
  run_conv<32, 64, 3, 3, 3, D2, H2, W2, D3, H3, W3, 2, 2, 2, 1, 1, 1, 0>(
      stream, mkp(buf2A, Wc3d, nullptr, d_in, 57, nullptr, mask3, buf3A, nullptr));
  resblock<64, D3, H3, W3>(d_in, stream, 62, buf3A, buf3B, Wc3w[0], Wc3w[1], mask3, nullptr);
  resblock<64, D3, H3, W3>(d_in, stream, 74, buf3A, buf3B, Wc3w[2], Wc3w[3], mask3, x3Out);

  maskdown(mask3, D3, H3, W3, mask4, D4, H4, W4, 3, 3, 3, 2, 2, 2, 0, 1, 1);
  run_conv<64, 128, 3, 3, 3, D3, H3, W3, D4, H4, W4, 2, 2, 2, 0, 1, 1, 0>(
      stream, mkp(buf3A, Wc4d, nullptr, d_in, 86, nullptr, mask4, buf4A, nullptr));
  resblock<128, D4, H4, W4>(d_in, stream, 91,  buf4A, buf4B, Wc4w[0], Wc4w[1], mask4, nullptr);
  resblock<128, D4, H4, W4>(d_in, stream, 103, buf4A, buf4B, Wc4w[2], Wc4w[3], mask4, x4Out);

  maskdown(mask4, D4, H4, W4, maskex, De, He, We, 3, 1, 1, 2, 1, 1, 0, 0, 0);
  run_conv<128, 128, 3, 1, 1, D4, H4, W4, De, He, We, 2, 1, 1, 0, 0, 0, 1>(
      stream, mkp(buf4A, Wex, nullptr, d_in, 115, nullptr, maskex, nullptr, retOut));
}
